// CustomGRU_40046275068070
// MI455X (gfx1250) — matrix operand order repaired and checked by reading, not compiled, not run on hardware
//
#include <hip/hip_runtime.h>

// ---------------------------------------------------------------------------
// 2-layer GRU, S=512, B=64, IN=H=1024.  bf16 WMMA (16x16x32) with f32 acc.
// gi GEMM uses async global->LDS staging (double buffered) for the shared
// weight tile; both GEMMs are software-pipelined in registers.
// ---------------------------------------------------------------------------

#define S_LEN  512
#define BATCH  64
#define INPUT  1024
#define HID    1024
#define LAYERS 2
#define NG3    (3 * HID)        // 3072
#define MROWS  (S_LEN * BATCH)  // 32768

typedef __attribute__((ext_vector_type(16))) __bf16 v16bf;
typedef __attribute__((ext_vector_type(8)))  __bf16 v8bf;
typedef __attribute__((ext_vector_type(8)))  float  v8f;

// Build a 16x32 bf16 A-fragment for one lane.
// Layout: lanes 0-15 = rows M=0..15, elems 0..15 = K[k+16*half..k+16*half+15],
// half = lane>>4 (contiguous, matches B fragments).
__device__ __forceinline__ v16bf make_a_frag(const __bf16* __restrict__ row,
                                             int k, int half) {
    v8bf lo = *reinterpret_cast<const v8bf*>(row + k + half * 16);
    v8bf hi = *reinterpret_cast<const v8bf*>(row + k + half * 16 + 8);
    v16bf a;
#pragma unroll
    for (int i = 0; i < 8; ++i) { a[i] = lo[i]; a[i + 8] = hi[i]; }
    return a;
}

// Async DMA one 64(row) x 32(k) bf16 weight tile into LDS (4 KB).
// 256 threads: thread t moves 16 bytes: row = t>>2, 16B chunk = t&3.
__device__ __forceinline__ void stage_b_async(const __bf16* __restrict__ W,
                                              int nbase, int k, int tid,
                                              __bf16* lds_tile) {
    const int row   = tid >> 2;
    const int chunk = tid & 3;
    const __bf16* src = W + (size_t)(nbase + row) * INPUT + k + chunk * 8;
    unsigned dst = (unsigned)(uintptr_t)(lds_tile + row * 32 + chunk * 8);
    asm volatile("global_load_async_to_lds_b128 %0, %1, off"
                 :: "v"(dst), "v"(src) : "memory");
}

// ---------------------------------------------------------------------------
// gi = X(bf16)[M,K] * W(bf16)[N,K]^T  -> GI(f32)[M,N]   (M=32768,K=1024,N=3072)
// Block: 256 thr = 8 waves; block tile 128(M) x 64(N); wave tile 16x64.
// B tile shared via async-LDS double buffer; A pipelined in registers.
// ---------------------------------------------------------------------------
__global__ void __launch_bounds__(256)
gi_gemm_kernel(const __bf16* __restrict__ X,
               const __bf16* __restrict__ W,
               float* __restrict__ GI) {
    __shared__ __bf16 btile[2][64][32];   // 2 x 4 KB double buffer

    const int tid   = threadIdx.x;
    const int lane  = tid & 31;
    const int wave  = tid >> 5;
    const int half  = lane >> 4;
    const int ln15  = lane & 15;
    const int nbase = blockIdx.x * 64;
    const int mbase = blockIdx.y * 128 + wave * 16;

    const __bf16* xrow = X + (size_t)(mbase + ln15) * INPUT;

    v8f acc0 = {}, acc1 = {}, acc2 = {}, acc3 = {};

    // Prologue: DMA first tile, prefetch first A fragment.
    stage_b_async(W, nbase, 0, tid, &btile[0][0][0]);
    v16bf a_cur = make_a_frag(xrow, 0, half);

#pragma unroll 2
    for (int it = 0; it < INPUT / 32; ++it) {
        const int k = it * 32;
        // Own async DMA into the current buffer has landed.
        asm volatile("s_wait_asynccnt 0x0" ::: "memory");
        // Everyone's DMA landed; everyone finished reading the other buffer.
        __builtin_amdgcn_s_barrier();
        // Kick off next tile's DMA; it overlaps this iteration's WMMAs.
        if (k + 32 < INPUT)
            stage_b_async(W, nbase, k + 32, tid, &btile[(it + 1) & 1][0][0]);
        // Pipeline next A fragment.
        v16bf a_nxt = a_cur;
        if (k + 32 < INPUT) a_nxt = make_a_frag(xrow, k + 32, half);

        const __bf16* bt = &btile[it & 1][0][0];
        v16bf b0 = *reinterpret_cast<const v16bf*>(bt + (ln15)      * 32 + half * 16);
        v16bf b1 = *reinterpret_cast<const v16bf*>(bt + (16 + ln15) * 32 + half * 16);
        v16bf b2 = *reinterpret_cast<const v16bf*>(bt + (32 + ln15) * 32 + half * 16);
        v16bf b3 = *reinterpret_cast<const v16bf*>(bt + (48 + ln15) * 32 + half * 16);

        acc0 = __builtin_amdgcn_wmma_f32_16x16x32_bf16(false, a_cur, false, b0, (short)0, acc0, false, false);
        acc1 = __builtin_amdgcn_wmma_f32_16x16x32_bf16(false, a_cur, false, b1, (short)0, acc1, false, false);
        acc2 = __builtin_amdgcn_wmma_f32_16x16x32_bf16(false, a_cur, false, b2, (short)0, acc2, false, false);
        acc3 = __builtin_amdgcn_wmma_f32_16x16x32_bf16(false, a_cur, false, b3, (short)0, acc3, false, false);
        a_cur = a_nxt;
    }

    // D layout: VGPR i -> M = 8*half + i, N = ln15.  GI is written once and
    // consumed much later -> nontemporal stores keep it out of L2.
    float* g0 = GI + (size_t)(mbase + half * 8) * NG3 + nbase + ln15;
#pragma unroll
    for (int i = 0; i < 8; ++i) {
        __builtin_nontemporal_store(acc0[i], g0 + (size_t)i * NG3 + 0);
        __builtin_nontemporal_store(acc1[i], g0 + (size_t)i * NG3 + 16);
        __builtin_nontemporal_store(acc2[i], g0 + (size_t)i * NG3 + 32);
        __builtin_nontemporal_store(acc3[i], g0 + (size_t)i * NG3 + 48);
    }
}

// ---------------------------------------------------------------------------
// One GRU timestep, fused: gh = h_{t-1}(bf16) * Wh^T  + gates + h update.
// Grid: 64 blocks (one per 16-wide slice of H). Block: 384 thr = 12 waves,
// wave (g,m) computes gate g, batch-tile m -> all 3 gate strips live in LDS.
// Register-pipelined K loop (h and Wh are L2-resident).
// ---------------------------------------------------------------------------
__global__ void __launch_bounds__(384)
gru_step_kernel(const __bf16* __restrict__ Hb,     // [B,H] bf16 h_{t-1}
                __bf16*       __restrict__ HbNext, // [B,H] bf16 h_t
                float*        __restrict__ Hf,     // [B,H] f32 h (in/out)
                const __bf16* __restrict__ Wh,     // [3H,H] bf16
                const float*  __restrict__ GIs,    // [B,3H] gi at this step
                const float*  __restrict__ bi,     // [3H]
                const float*  __restrict__ bh,     // [3H]
                float*        __restrict__ Out) {  // [B,H] layer output at s
    __shared__ float ghs[3][BATCH][16];

    const int lane = threadIdx.x & 31;
    const int wave = threadIdx.x >> 5;   // 0..11
    const int g    = wave >> 2;          // gate 0..2
    const int m    = wave & 3;           // batch tile 0..3
    const int half = lane >> 4;
    const int ln15 = lane & 15;
    const int n0   = blockIdx.x * 16;

    const __bf16* arow = Hb + (size_t)(m * 16 + ln15) * HID;
    const __bf16* brow = Wh + (size_t)(g * HID + n0 + ln15) * HID;

    v8f acc = {};
    v16bf a_cur = make_a_frag(arow, 0, half);
    v16bf b_cur = *reinterpret_cast<const v16bf*>(brow + half * 16);

#pragma unroll 4
    for (int k = 0; k < HID; k += 32) {
        v16bf a_nxt = a_cur, b_nxt = b_cur;
        if (k + 32 < HID) {
            a_nxt = make_a_frag(arow, k + 32, half);
            b_nxt = *reinterpret_cast<const v16bf*>(brow + k + 32 + half * 16);
        }
        acc = __builtin_amdgcn_wmma_f32_16x16x32_bf16(false, a_cur, false, b_cur, (short)0, acc, false, false);
        a_cur = a_nxt;
        b_cur = b_nxt;
    }

#pragma unroll
    for (int i = 0; i < 8; ++i)
        ghs[g][m * 16 + half * 8 + i][ln15] = acc[i];
    __syncthreads();

    for (int idx = threadIdx.x; idx < BATCH * 16; idx += 384) {
        const int b   = idx >> 4;
        const int n   = idx & 15;
        const int col = n0 + n;
        const float gh_r = ghs[0][b][n] + bh[col];
        const float gh_z = ghs[1][b][n] + bh[HID + col];
        const float gh_n = ghs[2][b][n] + bh[2 * HID + col];
        const float* gip = GIs + (size_t)b * NG3;
        const float gi_r = __builtin_nontemporal_load(gip + col)           + bi[col];
        const float gi_z = __builtin_nontemporal_load(gip + HID + col)     + bi[HID + col];
        const float gi_n = __builtin_nontemporal_load(gip + 2 * HID + col) + bi[2 * HID + col];
        const float r  = 1.f / (1.f + __expf(-(gi_r + gh_r)));
        const float z  = 1.f / (1.f + __expf(-(gi_z + gh_z)));
        const float nn = tanhf(gi_n + r * gh_n);
        const float hp = Hf[(size_t)b * HID + col];
        const float hn = (1.f - z) * nn + z * hp;
        Hf[(size_t)b * HID + col]     = hn;
        HbNext[(size_t)b * HID + col] = (__bf16)hn;
        Out[(size_t)b * HID + col]    = hn;
    }
}

// ---------------------------------------------------------------------------
// Utility kernels
// ---------------------------------------------------------------------------
__global__ void cast_f32_bf16_kernel(const float* __restrict__ src,
                                     __bf16* __restrict__ dst, int n) {
    int i = (blockIdx.x * blockDim.x + threadIdx.x) * 4;
    if (i + 3 < n) {
        float4 v = *reinterpret_cast<const float4*>(src + i);
        dst[i]     = (__bf16)v.x;
        dst[i + 1] = (__bf16)v.y;
        dst[i + 2] = (__bf16)v.z;
        dst[i + 3] = (__bf16)v.w;
    } else {
        for (; i < n; ++i) dst[i] = (__bf16)src[i];
    }
}

__global__ void h_init_kernel(const float* __restrict__ h0,
                              float* __restrict__ Hf,
                              __bf16* __restrict__ Hb) {
    int i = blockIdx.x * blockDim.x + threadIdx.x;
    if (i < BATCH * HID) {
        float v = h0[i];
        Hf[i] = v;
        Hb[i] = (__bf16)v;
    }
}

// ---------------------------------------------------------------------------
// Launcher
// ---------------------------------------------------------------------------
extern "C" void kernel_launch(void* const* d_in, const int* in_sizes, int n_in,
                              void* d_out, int out_size, void* d_ws, size_t ws_size,
                              hipStream_t stream) {
    const float* x  = (const float*)d_in[0];   // [S,B,IN]
    const float* h0 = (const float*)d_in[1];   // [L,B,H]
    const float* Wi = (const float*)d_in[2];   // [L,3,H,IN]
    const float* Wh = (const float*)d_in[3];   // [L,3,H,H]
    const float* bi = (const float*)d_in[4];   // [L,3,H]
    const float* bh = (const float*)d_in[5];   // [L,3,H]

    float* out_seq = (float*)d_out;                          // [S,B,H]
    float* out_hn  = out_seq + (size_t)S_LEN * BATCH * HID;  // [L,B,H]

    // Workspace carve-up (256B aligned slices).
    char* w = (char*)d_ws;
    auto take = [&](size_t bytes) -> char* {
        char* p = w;
        w += (bytes + 255) & ~(size_t)255;
        return p;
    };
    __bf16* xb   = (__bf16*)take((size_t)MROWS * INPUT * 2);            // 64 MB
    __bf16* Wib  = (__bf16*)take((size_t)LAYERS * 3 * HID * INPUT * 2); // 12.6 MB
    __bf16* Whb  = (__bf16*)take((size_t)LAYERS * 3 * HID * HID * 2);   // 12.6 MB
    float*  GI   = (float*)take((size_t)MROWS * NG3 * 4);               // 403 MB
    float*  seq0 = (float*)take((size_t)MROWS * HID * 4);               // 134 MB
    float*  Hf   = (float*)take((size_t)BATCH * HID * 4);
    __bf16* Hb0  = (__bf16*)take((size_t)BATCH * HID * 2);
    __bf16* Hb1  = (__bf16*)take((size_t)BATCH * HID * 2);

    // Cast weights to bf16 once per call.
    {
        int n = LAYERS * 3 * HID * INPUT;
        cast_f32_bf16_kernel<<<(n / 4 + 255) / 256, 256, 0, stream>>>(Wi, Wib, n);
        n = LAYERS * 3 * HID * HID;
        cast_f32_bf16_kernel<<<(n / 4 + 255) / 256, 256, 0, stream>>>(Wh, Whb, n);
    }

    for (int l = 0; l < LAYERS; ++l) {
        const float* in_seq = (l == 0) ? x : seq0;
        float*       o      = (l == LAYERS - 1) ? out_seq : seq0;

        // Cast this layer's input sequence to bf16.
        int n = MROWS * INPUT;
        cast_f32_bf16_kernel<<<(n / 4 + 255) / 256, 256, 0, stream>>>(in_seq, xb, n);

        // Big parallel GEMM: all input-side gate projections for the layer.
        gi_gemm_kernel<<<dim3(NG3 / 64, MROWS / 128), 256, 0, stream>>>(
            xb, Wib + (size_t)l * 3 * HID * INPUT, GI);

        // h_0 for this layer.
        h_init_kernel<<<(BATCH * HID + 255) / 256, 256, 0, stream>>>(
            h0 + (size_t)l * BATCH * HID, Hf, Hb0);

        // Sequential scan: one fused kernel per timestep (graph-replayed).
        for (int s = 0; s < S_LEN; ++s) {
            const __bf16* hin  = (s & 1) ? Hb1 : Hb0;
            __bf16*       hout = (s & 1) ? Hb0 : Hb1;
            gru_step_kernel<<<HID / 16, 384, 0, stream>>>(
                hin, hout, Hf,
                Whb + (size_t)l * 3 * HID * HID,
                GI + (size_t)s * BATCH * NG3,
                bi + (size_t)l * 3 * HID,
                bh + (size_t)l * 3 * HID,
                o + (size_t)s * BATCH * HID);
        }

        // Final hidden state for this layer -> d_out h_n slot.
        hipMemcpyAsync(out_hn + (size_t)l * BATCH * HID, Hf,
                       (size_t)BATCH * HID * sizeof(float),
                       hipMemcpyDeviceToDevice, stream);
    }
}
